// WETE_20426864460398
// MI455X (gfx1250) — matrix operand order repaired and checked by reading, not compiled, not run on hardware
//
#include <hip/hip_runtime.h>
#include <hip/hip_bf16.h>
#include <math.h>

#define NB   512     // docs
#define NV   50000   // vocab
#define NK   200     // topics
#define NH   300     // embed dim
#define NHID 800     // hidden

typedef __bf16 v16bf __attribute__((ext_vector_type(16)));
typedef float  v8f   __attribute__((ext_vector_type(8)));

__device__ __forceinline__ int imin(int a, int b) { return a < b ? a : b; }
__device__ __forceinline__ float clipv(float e) {
  return fminf(fmaxf(e, 1e-30f), 1e10f);
}

__device__ __forceinline__ v8f wmma_bf16(v16bf a, v16bf b, v8f c) {
  return __builtin_amdgcn_wmma_f32_16x16x32_bf16(false, a, false, b, (short)0, c,
                                                 false, false);
}

__device__ __forceinline__ void cvt4(v16bf& a, int base, float4 u) {
  a[base + 0] = (__bf16)u.x; a[base + 1] = (__bf16)u.y;
  a[base + 2] = (__bf16)u.z; a[base + 3] = (__bf16)u.w;
}

// ---------------------------------------------------------------------------
// A fragment, row-major source contiguous in k.  lane = kh*16 + m.
// element e -> k = (e/8)*16 + kh*8 + (e%8): two contiguous 8-float runs.
// ---------------------------------------------------------------------------
__device__ __forceinline__ v16bf load_a_rm(const float* rowk, int kh) {
  const float* p = rowk + kh * 8;
  v16bf a;
  cvt4(a, 0,  *(const float4*)(p));
  cvt4(a, 4,  *(const float4*)(p + 4));
  cvt4(a, 8,  *(const float4*)(p + 16));
  cvt4(a, 12, *(const float4*)(p + 20));
  return a;
}
// Tail with 16 valid k: only g=0 elements live, first 16 k are in-bounds.
__device__ __forceinline__ v16bf load_a_tail16(const float* rowk, int kh) {
  const float* p = rowk + kh * 8;
  v16bf a;
  cvt4(a, 0, *(const float4*)(p));
  cvt4(a, 4, *(const float4*)(p + 4));
#pragma unroll
  for (int e = 8; e < 16; ++e) a[e] = (__bf16)0.0f;
  return a;
}
// Tail with 8 valid k (only kh==0, e<8 live).  Loads are in-bounds for all lanes.
__device__ __forceinline__ v16bf load_a_tail8(const float* rowk, int kh) {
  float4 u0 = *(const float4*)(rowk);
  float4 u1 = *(const float4*)(rowk + 4);
  const bool z = (kh != 0);
  v16bf a;
  a[0] = (__bf16)(z ? 0.f : u0.x); a[1] = (__bf16)(z ? 0.f : u0.y);
  a[2] = (__bf16)(z ? 0.f : u0.z); a[3] = (__bf16)(z ? 0.f : u0.w);
  a[4] = (__bf16)(z ? 0.f : u1.x); a[5] = (__bf16)(z ? 0.f : u1.y);
  a[6] = (__bf16)(z ? 0.f : u1.z); a[7] = (__bf16)(z ? 0.f : u1.w);
#pragma unroll
  for (int e = 8; e < 16; ++e) a[e] = (__bf16)0.0f;
  return a;
}
// Tail with 12 valid k (NH=300): kh=0 -> e0..7 valid; kh=1 -> e0..3 valid.
__device__ __forceinline__ v16bf load_a_tail12(const float* rowk, int kh) {
  float4 u0 = *(const float4*)(rowk + kh * 8);
  float4 u1 = *(const float4*)(rowk + (kh ? 8 : 4));   // clamped re-read for kh=1
  const bool z = (kh != 0);
  v16bf a;
  a[0] = (__bf16)u0.x; a[1] = (__bf16)u0.y; a[2] = (__bf16)u0.z; a[3] = (__bf16)u0.w;
  a[4] = (__bf16)(z ? 0.f : u1.x); a[5] = (__bf16)(z ? 0.f : u1.y);
  a[6] = (__bf16)(z ? 0.f : u1.z); a[7] = (__bf16)(z ? 0.f : u1.w);
#pragma unroll
  for (int e = 8; e < 16; ++e) a[e] = (__bf16)0.0f;
  return a;
}

// ---------------------------------------------------------------------------
// B fragment from a source contiguous in k for fixed n (transposed views /
// LDS [n][k] tiles).  lane = kh*16 + n.  element e -> k = (e/8)*16 + kh*8 + (e%8).
// ---------------------------------------------------------------------------
__device__ __forceinline__ v16bf load_bT_rm(const float* colk, int kh) {
  const float* p = colk + kh * 8;
  v16bf b;
  cvt4(b, 0,  *(const float4*)(p));
  cvt4(b, 4,  *(const float4*)(p + 4));
  cvt4(b, 8,  *(const float4*)(p + 16));
  cvt4(b, 12, *(const float4*)(p + 20));
  return b;
}

// ---------------------------------------------------------------------------
// Kernel 0: zero accumulators + scale pad (re-run every launch)
// ---------------------------------------------------------------------------
__global__ void k_init(float* __restrict__ fdoc, float* __restrict__ scale,
                       float* __restrict__ acc) {
  int i = blockIdx.x * blockDim.x + threadIdx.x;
  if (i < NB) fdoc[i] = 0.0f;
  if (i < 4)  acc[i]  = 0.0f;
  if (i >= NK && i < 224) scale[i] = 0.0f;   // k-pad for vectorized phi loads
}

// ---------------------------------------------------------------------------
// Kernel 1: n_tok[b] = sum_v bows[b,v]
// ---------------------------------------------------------------------------
__global__ void k_ntok(const float* __restrict__ bows, float* __restrict__ ntok) {
  __shared__ float red[256];
  const int b = blockIdx.x, t = threadIdx.x;
  float s = 0.0f;
  for (int v = t; v < NV; v += 256) s += bows[b * NV + v];
  red[t] = s; __syncthreads();
  for (int off = 128; off > 0; off >>= 1) {
    if (t < off) red[t] += red[t + off];
    __syncthreads();
  }
  if (t == 0) ntok[b] = red[0];
}

// ---------------------------------------------------------------------------
// Kernel 2: hid = relu(bows @ W1 + b1)   [512 x 50000] @ [50000 x 800]
// block 128 (4 waves); wave 32x32; block tile 128x32; B staged via LDS.
// K-tail peeled so the 1562-iteration steady state is branch/clamp free.
// ---------------------------------------------------------------------------
__global__ void k_gemm_hid(const float* __restrict__ bows,
                           const float* __restrict__ W1,
                           const float* __restrict__ b1,
                           float* __restrict__ hid) {
  __shared__ __align__(16) float lb[32][36];          // [n][k], stride 36
  const int tid = threadIdx.x;
  const int lane = tid & 31, wave = tid >> 5;
  const int kh = lane >> 4, nl = lane & 15;
  const int n0 = blockIdx.x * 32;
  const int m0 = blockIdx.y * 128 + wave * 32;
  const float* arow0 = bows + (m0 + nl) * NV;
  const float* arow1 = arow0 + 16 * NV;
  const int sr = tid >> 2, sc0 = (tid & 3) * 8;       // staging coords

  v8f acc[2][2] = {};
  int kb = 0;
  for (; kb + 32 <= NV; kb += 32) {                   // kb+sr <= 49983: in-bounds
    const float* src = W1 + (kb + sr) * NHID + n0 + sc0;
    float4 x0 = *(const float4*)src;
    float4 x1 = *(const float4*)(src + 4);
    __builtin_prefetch(W1 + imin(kb + sr + 32, NV - 1) * NHID + n0 + sc0, 0, 1);
    __syncthreads();
    lb[sc0 + 0][sr] = x0.x; lb[sc0 + 1][sr] = x0.y;
    lb[sc0 + 2][sr] = x0.z; lb[sc0 + 3][sr] = x0.w;
    lb[sc0 + 4][sr] = x1.x; lb[sc0 + 5][sr] = x1.y;
    lb[sc0 + 6][sr] = x1.z; lb[sc0 + 7][sr] = x1.w;
    __syncthreads();
    v16bf a0 = load_a_rm(arow0 + kb, kh);
    v16bf a1 = load_a_rm(arow1 + kb, kh);
    v16bf bv0 = load_bT_rm(&lb[nl][0], kh);
    v16bf bv1 = load_bT_rm(&lb[nl + 16][0], kh);
    acc[0][0] = wmma_bf16(a0, bv0, acc[0][0]);
    acc[0][1] = wmma_bf16(a0, bv1, acc[0][1]);
    acc[1][0] = wmma_bf16(a1, bv0, acc[1][0]);
    acc[1][1] = wmma_bf16(a1, bv1, acc[1][1]);
  }
  { // peeled tail: kb = 49984, 16 valid k rows (row-clamped staging)
    const float* src = W1 + imin(kb + sr, NV - 1) * NHID + n0 + sc0;
    float4 x0 = *(const float4*)src;
    float4 x1 = *(const float4*)(src + 4);
    __syncthreads();
    lb[sc0 + 0][sr] = x0.x; lb[sc0 + 1][sr] = x0.y;
    lb[sc0 + 2][sr] = x0.z; lb[sc0 + 3][sr] = x0.w;
    lb[sc0 + 4][sr] = x1.x; lb[sc0 + 5][sr] = x1.y;
    lb[sc0 + 6][sr] = x1.z; lb[sc0 + 7][sr] = x1.w;
    __syncthreads();
    v16bf a0 = load_a_tail16(arow0 + kb, kh);
    v16bf a1 = load_a_tail16(arow1 + kb, kh);
    v16bf bv0 = load_bT_rm(&lb[nl][0], kh);
    v16bf bv1 = load_bT_rm(&lb[nl + 16][0], kh);
    acc[0][0] = wmma_bf16(a0, bv0, acc[0][0]);
    acc[0][1] = wmma_bf16(a0, bv1, acc[0][1]);
    acc[1][0] = wmma_bf16(a1, bv0, acc[1][0]);
    acc[1][1] = wmma_bf16(a1, bv1, acc[1][1]);
  }

  const int mh = kh * 8;
#pragma unroll
  for (int i = 0; i < 2; ++i)
#pragma unroll
    for (int j = 0; j < 2; ++j) {
      const int n = n0 + j * 16 + nl;
      const float bias = b1[n];
#pragma unroll
      for (int r = 0; r < 8; ++r) {
        const int m = m0 + i * 16 + mh + r;
        const float v = acc[i][j][r] + bias;
        hid[m * NHID + n] = v > 0.0f ? v : 0.0f;
      }
    }
}

// ---------------------------------------------------------------------------
// Kernel 3: theta = softplus(hid @ W2 + b2)   (small, VALU)
// ---------------------------------------------------------------------------
__global__ void k_theta(const float* __restrict__ hid,
                        const float* __restrict__ W2,
                        const float* __restrict__ b2,
                        float* __restrict__ theta) {
  const int idx = blockIdx.x * blockDim.x + threadIdx.x;
  if (idx >= NB * NK) return;
  const int b = idx / NK, k = idx % NK;
  float s = b2[k];
  const float* h = hid + b * NHID;
  for (int j = 0; j < NHID; j += 4) {
    float4 hv = *(const float4*)(h + j);
    s = fmaf(hv.x, W2[(j + 0) * NK + k], s);
    s = fmaf(hv.y, W2[(j + 1) * NK + k], s);
    s = fmaf(hv.z, W2[(j + 2) * NK + k], s);
    s = fmaf(hv.w, W2[(j + 3) * NK + k], s);
  }
  theta[idx] = (s > 20.0f) ? s : log1pf(expf(s));
}

// ---------------------------------------------------------------------------
// Kernel 4: theta_norm = softmax(theta, axis=1)
// ---------------------------------------------------------------------------
__global__ void k_softmax_rows(const float* __restrict__ theta,
                               float* __restrict__ tnorm) {
  __shared__ float red[256];
  const int b = blockIdx.x, t = threadIdx.x;
  float m = -1e30f;
  for (int k = t; k < NK; k += 256) m = fmaxf(m, theta[b * NK + k]);
  red[t] = m; __syncthreads();
  for (int off = 128; off > 0; off >>= 1) {
    if (t < off) red[t] = fmaxf(red[t], red[t + off]);
    __syncthreads();
  }
  m = red[0]; __syncthreads();
  float s = 0.0f;
  for (int k = t; k < NK; k += 256) s += __expf(theta[b * NK + k] - m);
  red[t] = s; __syncthreads();
  for (int off = 128; off > 0; off >>= 1) {
    if (t < off) red[t] += red[t + off];
    __syncthreads();
  }
  s = red[0];
  for (int k = t; k < NK; k += 256) tnorm[b * NK + k] = __expf(theta[b * NK + k] - m) / s;
}

// ---------------------------------------------------------------------------
// Kernel 5: inner = rho @ alpha.T   [50000 x 300] @ [300 x 200]
// ---------------------------------------------------------------------------
__global__ void k_gemm_inner(const float* __restrict__ rho,
                             const float* __restrict__ alpha,
                             float* __restrict__ inner) {
  const int lane = threadIdx.x & 31, wave = threadIdx.x >> 5;
  const int kh = lane >> 4, nl = lane & 15;
  const int n0 = blockIdx.x * 32;
  const int m0 = blockIdx.y * 128 + wave * 32;
  const float* arow0 = rho + imin(m0 + nl, NV - 1) * NH;
  const float* arow1 = rho + imin(m0 + 16 + nl, NV - 1) * NH;
  const float* bcol0 = alpha + imin(n0 + nl, NK - 1) * NH;        // alpha.T col
  const float* bcol1 = alpha + imin(n0 + 16 + nl, NK - 1) * NH;

  v8f acc[2][2] = {};
  for (int kb = 0; kb < 288; kb += 32) {
    v16bf a0 = load_a_rm(arow0 + kb, kh);
    v16bf a1 = load_a_rm(arow1 + kb, kh);
    v16bf bv0 = load_bT_rm(bcol0 + kb, kh);   // contiguous in k (alpha rows)
    v16bf bv1 = load_bT_rm(bcol1 + kb, kh);
    acc[0][0] = wmma_bf16(a0, bv0, acc[0][0]);
    acc[0][1] = wmma_bf16(a0, bv1, acc[0][1]);
    acc[1][0] = wmma_bf16(a1, bv0, acc[1][0]);
    acc[1][1] = wmma_bf16(a1, bv1, acc[1][1]);
  }
  { // tail: 12 valid k (288..299); A zeroed past 300, B address-clamped
    v16bf a0 = load_a_tail12(arow0 + 288, kh);
    v16bf a1 = load_a_tail12(arow1 + 288, kh);
    v16bf bv0, bv1;
#pragma unroll
    for (int e = 0; e < 16; ++e) {
      const int kk = imin(288 + kh * 8 + (e & 7) + (e >> 3) * 16, NH - 1);
      bv0[e] = (__bf16)bcol0[kk];
      bv1[e] = (__bf16)bcol1[kk];
    }
    acc[0][0] = wmma_bf16(a0, bv0, acc[0][0]);
    acc[0][1] = wmma_bf16(a0, bv1, acc[0][1]);
    acc[1][0] = wmma_bf16(a1, bv0, acc[1][0]);
    acc[1][1] = wmma_bf16(a1, bv1, acc[1][1]);
  }

  const int mh = kh * 8;
#pragma unroll
  for (int i = 0; i < 2; ++i)
#pragma unroll
    for (int j = 0; j < 2; ++j) {
      const int n = n0 + j * 16 + nl;
      if (n >= NK) continue;
#pragma unroll
      for (int r = 0; r < 8; ++r) {
        const int m = m0 + i * 16 + mh + r;
        if (m < NV) inner[m * NK + n] = acc[i][j][r];
      }
    }
}

// ---------------------------------------------------------------------------
// Kernel 6: per-topic vocab softmax folded scale: scale[k]=exp(-max)/sumexp
// ---------------------------------------------------------------------------
__global__ void k_colred(const float* __restrict__ inner, float* __restrict__ scale) {
  __shared__ float red[256];
  const int k = blockIdx.x, t = threadIdx.x;
  float m = -1e30f;
  for (int v = t; v < NV; v += 256) m = fmaxf(m, inner[v * NK + k]);
  red[t] = m; __syncthreads();
  for (int off = 128; off > 0; off >>= 1) {
    if (t < off) red[t] = fmaxf(red[t], red[t + off]);
    __syncthreads();
  }
  m = red[0]; __syncthreads();
  float s = 0.0f;
  for (int v = t; v < NV; v += 256) s += __expf(inner[v * NK + k] - m);
  red[t] = s; __syncthreads();
  for (int off = 128; off > 0; off >>= 1) {
    if (t < off) red[t] += red[t + off];
    __syncthreads();
  }
  if (t == 0) scale[k] = __expf(-m) / red[0];
}

// ---------------------------------------------------------------------------
// Kernel 7: forward cost.  wave tile 16(v) x 64(docs); dis/cd built once per
// A tile from inner (vector loads), reused over 4 doc subtiles.
// ---------------------------------------------------------------------------
__global__ void k_fwd(const float* __restrict__ inner,
                      const float* __restrict__ tnorm,
                      const float* __restrict__ bows,
                      float* __restrict__ fdoc) {
  const int lane = threadIdx.x & 31, wave = threadIdx.x >> 5;
  const int kh = lane >> 4, nl = lane & 15;
  const int b0 = blockIdx.x * 64;
  const int v0 = blockIdx.y * 64 + wave * 16;
  const float* arow = inner + imin(v0 + nl, NV - 1) * NK;
  const float* brow[4];
#pragma unroll
  for (int j = 0; j < 4; ++j) brow[j] = tnorm + (b0 + j * 16 + nl) * NK;

  v8f num[4] = {}, den[4] = {};
  for (int kb = 0; kb < 192; kb += 32) {
    const float* p = arow + kb + kh * 8;
    float4 u0 = *(const float4*)(p),      u1 = *(const float4*)(p + 4);
    float4 u2 = *(const float4*)(p + 16), u3 = *(const float4*)(p + 20);
    float xs[16] = {u0.x,u0.y,u0.z,u0.w, u1.x,u1.y,u1.z,u1.w,
                    u2.x,u2.y,u2.z,u2.w, u3.x,u3.y,u3.z,u3.w};
    v16bf ad, ac;
#pragma unroll
    for (int e = 0; e < 16; ++e) {
      const float d = clipv(__expf(xs[e]));
      const float c = clipv(__expf(-xs[e]));
      ad[e] = (__bf16)d; ac[e] = (__bf16)(d * c);
    }
#pragma unroll
    for (int j = 0; j < 4; ++j) {
      v16bf bt = load_bT_rm(brow[j] + kb, kh);
      num[j] = wmma_bf16(ac, bt, num[j]);
      den[j] = wmma_bf16(ad, bt, den[j]);
    }
  }
  { // tail kb=192: 8 valid k (kh==0, e<8)
    const float* p = arow + 192 + kh * 8;
    float4 u0 = *(const float4*)(p), u1 = *(const float4*)(p + 4);
    float xs[8] = {u0.x,u0.y,u0.z,u0.w, u1.x,u1.y,u1.z,u1.w};
    v16bf ad, ac;
#pragma unroll
    for (int e = 0; e < 8; ++e) {
      const float d = clipv(__expf(xs[e]));
      const float c = clipv(__expf(-xs[e]));
      const bool ok = (kh == 0);
      ad[e] = (__bf16)(ok ? d : 0.f); ac[e] = (__bf16)(ok ? d * c : 0.f);
    }
#pragma unroll
    for (int e = 8; e < 16; ++e) { ad[e] = (__bf16)0.f; ac[e] = (__bf16)0.f; }
#pragma unroll
    for (int j = 0; j < 4; ++j) {
      v16bf bt = load_bT_rm(brow[j] + 192, kh);   // overruns row; killed by zero A
      num[j] = wmma_bf16(ac, bt, num[j]);
      den[j] = wmma_bf16(ad, bt, den[j]);
    }
  }

  const int mh = kh * 8;
#pragma unroll
  for (int j = 0; j < 4; ++j) {
    const int doc = b0 + j * 16 + nl;
    float part = 0.0f;
#pragma unroll
    for (int r = 0; r < 8; ++r) {
      const int v = v0 + mh + r;
      if (v < NV) part += bows[doc * NV + v] * num[j][r] / (den[j][r] + 1e-30f);
    }
    part += __shfl_xor(part, 16, 32);
    if (lane < 16) atomicAdd(&fdoc[doc], part);
  }
}

// ---------------------------------------------------------------------------
// Kernel 8: colsum = bows@dis, Abd = bows@cd.  B (dis/cd from inner) staged
// in LDS transposed; A = bows vector loads.  wave 16(doc)x16(topic).
// K-tail peeled: steady state branch/clamp free.
// ---------------------------------------------------------------------------
__global__ void k_colsumA(const float* __restrict__ bows,
                          const float* __restrict__ inner,
                          float* __restrict__ colsum,
                          float* __restrict__ Abd) {
  __shared__ __align__(16) float lS[16][36];
  __shared__ __align__(16) float lA[16][36];
  const int tid = threadIdx.x;
  const int lane = tid & 31, wave = tid >> 5;
  const int kh = lane >> 4, nl = lane & 15;
  const int n0 = blockIdx.x * 16;
  const int m0 = blockIdx.y * 64 + wave * 16;
  const float* arow = bows + (m0 + nl) * NV;
  const int sr = tid >> 2, sc = (tid & 3) * 4;
  const int scc = imin(n0 + sc, NK - 4);

  v8f accS = {}, accA = {};
  int kb = 0;
  for (; kb + 32 <= NV; kb += 32) {                   // kb+sr <= 49983: in-bounds
    float4 x = *(const float4*)(inner + (kb + sr) * NK + scc);
    float xs[4] = {x.x, x.y, x.z, x.w};
    __syncthreads();
#pragma unroll
    for (int u = 0; u < 4; ++u) {
      const float d = clipv(__expf(xs[u]));
      const float c = d * clipv(__expf(-xs[u]));
      lS[sc + u][sr] = d;
      lA[sc + u][sr] = c;
    }
    __syncthreads();
    v16bf a  = load_a_rm(arow + kb, kh);
    v16bf bS = load_bT_rm(&lS[nl][0], kh);
    v16bf bA = load_bT_rm(&lA[nl][0], kh);
    accS = wmma_bf16(a, bS, accS);
    accA = wmma_bf16(a, bA, accA);
  }
  { // peeled tail: kb = 49984, 16 valid rows (row-clamped staging)
    float4 x = *(const float4*)(inner + imin(kb + sr, NV - 1) * NK + scc);
    float xs[4] = {x.x, x.y, x.z, x.w};
    __syncthreads();
#pragma unroll
    for (int u = 0; u < 4; ++u) {
      const float d = clipv(__expf(xs[u]));
      const float c = d * clipv(__expf(-xs[u]));
      lS[sc + u][sr] = d;
      lA[sc + u][sr] = c;
    }
    __syncthreads();
    v16bf a  = load_a_tail16(arow + kb, kh);
    v16bf bS = load_bT_rm(&lS[nl][0], kh);
    v16bf bA = load_bT_rm(&lA[nl][0], kh);
    accS = wmma_bf16(a, bS, accS);
    accA = wmma_bf16(a, bA, accA);
  }

  const int n = n0 + nl;
  if (n < NK) {
    const int mh = kh * 8;
#pragma unroll
    for (int r = 0; r < 8; ++r) {
      const int m = m0 + mh + r;
      colsum[m * NK + n] = accS[r];
      Abd[m * NK + n]    = accA[r];
    }
  }
}

// ---------------------------------------------------------------------------
// Kernel 9: Poisson reconstruction.  wave tile 64(docs) x 16(v); one phi
// fragment (exp(x)*scale[k], vector loads) reused over 4 doc subtiles.
// ---------------------------------------------------------------------------
__global__ void k_recon(const float* __restrict__ theta,
                        const float* __restrict__ inner,
                        const float* __restrict__ scale,
                        const float* __restrict__ bows,
                        float* __restrict__ acc) {
  const int lane = threadIdx.x & 31, wave = threadIdx.x >> 5;
  const int kh = lane >> 4, nl = lane & 15;
  const int m0 = blockIdx.y * 64;                    // docs
  const int v0 = blockIdx.x * 64 + wave * 16;        // vocab
  const float* arow[4];
#pragma unroll
  for (int i = 0; i < 4; ++i) arow[i] = theta + (m0 + i * 16 + nl) * NK;
  const float* brow = inner + imin(v0 + nl, NV - 1) * NK;

  v8f d[4] = {};
  for (int kb = 0; kb < 192; kb += 32) {
    const float* p = brow + kb + kh * 8;
    const float* q = scale + kb + kh * 8;
    float4 u0 = *(const float4*)(p),     u1 = *(const float4*)(p + 4);
    float4 u2 = *(const float4*)(p + 16), u3 = *(const float4*)(p + 20);
    float4 s0 = *(const float4*)(q),     s1 = *(const float4*)(q + 4);
    float4 s2 = *(const float4*)(q + 16), s3 = *(const float4*)(q + 20);
    float xs[16] = {u0.x,u0.y,u0.z,u0.w, u1.x,u1.y,u1.z,u1.w,
                    u2.x,u2.y,u2.z,u2.w, u3.x,u3.y,u3.z,u3.w};
    float ss[16] = {s0.x,s0.y,s0.z,s0.w, s1.x,s1.y,s1.z,s1.w,
                    s2.x,s2.y,s2.z,s2.w, s3.x,s3.y,s3.z,s3.w};
    v16bf bphi;
#pragma unroll
    for (int e = 0; e < 16; ++e) bphi[e] = (__bf16)(__expf(xs[e]) * ss[e]);
    v16bf a0 = load_a_rm(arow[0] + kb, kh);
    v16bf a1 = load_a_rm(arow[1] + kb, kh);
    v16bf a2 = load_a_rm(arow[2] + kb, kh);
    v16bf a3 = load_a_rm(arow[3] + kb, kh);
    d[0] = wmma_bf16(a0, bphi, d[0]);
    d[1] = wmma_bf16(a1, bphi, d[1]);
    d[2] = wmma_bf16(a2, bphi, d[2]);
    d[3] = wmma_bf16(a3, bphi, d[3]);
  }
  { // peeled tail kb=192: scale pad zeroes k>=200 in B; A zeroed past 8 valid k
    const float* p = brow + 192 + kh * 8;
    const float* q = scale + 192 + kh * 8;
    float4 u0 = *(const float4*)(p),     u1 = *(const float4*)(p + 4);
    float4 u2 = *(const float4*)(p + 16), u3 = *(const float4*)(p + 20);
    float4 s0 = *(const float4*)(q),     s1 = *(const float4*)(q + 4);
    float4 s2 = *(const float4*)(q + 16), s3 = *(const float4*)(q + 20);
    float xs[16] = {u0.x,u0.y,u0.z,u0.w, u1.x,u1.y,u1.z,u1.w,
                    u2.x,u2.y,u2.z,u2.w, u3.x,u3.y,u3.z,u3.w};
    float ss[16] = {s0.x,s0.y,s0.z,s0.w, s1.x,s1.y,s1.z,s1.w,
                    s2.x,s2.y,s2.z,s2.w, s3.x,s3.y,s3.z,s3.w};
    v16bf bphi;
#pragma unroll
    for (int e = 0; e < 16; ++e) bphi[e] = (__bf16)(__expf(xs[e]) * ss[e]);
    v16bf a0 = load_a_tail8(arow[0] + 192, kh);
    v16bf a1 = load_a_tail8(arow[1] + 192, kh);
    v16bf a2 = load_a_tail8(arow[2] + 192, kh);
    v16bf a3 = load_a_tail8(arow[3] + 192, kh);
    d[0] = wmma_bf16(a0, bphi, d[0]);
    d[1] = wmma_bf16(a1, bphi, d[1]);
    d[2] = wmma_bf16(a2, bphi, d[2]);
    d[3] = wmma_bf16(a3, bphi, d[3]);
  }

  const int v = v0 + nl;
  const int mh = kh * 8;
  float part = 0.0f;
  if (v < NV) {
#pragma unroll
    for (int i = 0; i < 4; ++i)
#pragma unroll
      for (int r = 0; r < 8; ++r) {
        const int doc = m0 + i * 16 + mh + r;
        const float rc = d[i][r];
        const float c = bows[doc * NV + v];
        part += -(c * __logf(rc + 1e-10f) - rc - lgammaf(c + 1.0f));
      }
  }
  for (int off = 16; off > 0; off >>= 1) part += __shfl_xor(part, off, 32);
  if (lane == 0) atomicAdd(&acc[2], part);
}

// ---------------------------------------------------------------------------
// Kernel 10: per-doc finals -> acc[0] (forward), acc[1] (backward)
// ---------------------------------------------------------------------------
__global__ void k_docfinal(const float* __restrict__ Abd,
                           const float* __restrict__ colsum,
                           const float* __restrict__ tnorm,
                           const float* __restrict__ ntok,
                           const float* __restrict__ fdoc,
                           float* __restrict__ acc) {
  __shared__ float red[256];
  const int b = blockIdx.x, t = threadIdx.x;
  float s = 0.0f;
  for (int k = t; k < NK; k += 256)
    s += Abd[b * NK + k] / (colsum[b * NK + k] + 1e-30f) * tnorm[b * NK + k];
  red[t] = s; __syncthreads();
  for (int off = 128; off > 0; off >>= 1) {
    if (t < off) red[t] += red[t + off];
    __syncthreads();
  }
  if (t == 0) {
    const float nt = ntok[b];
    if (nt > 0.0f) {
      atomicAdd(&acc[1], red[0]);
      atomicAdd(&acc[0], fdoc[b] / fmaxf(nt, 1.0f));
    }
  }
}

__global__ void k_final(const float* __restrict__ acc, float* __restrict__ out) {
  out[0] = 1.0f * (acc[2] / (float)NB);   // EPSILON * tm
  out[1] = 0.5f * acc[0];                 // BETA * forward
  out[2] = 0.5f * acc[1];                 // (1-BETA) * backward
}

// ---------------------------------------------------------------------------
// Launch
// ---------------------------------------------------------------------------
extern "C" void kernel_launch(void* const* d_in, const int* in_sizes, int n_in,
                              void* d_out, int out_size, void* d_ws, size_t ws_size,
                              hipStream_t stream) {
  const float* bows  = (const float*)d_in[0];
  const float* rho   = (const float*)d_in[1];
  const float* alpha = (const float*)d_in[2];
  const float* W1    = (const float*)d_in[3];
  const float* b1    = (const float*)d_in[4];
  const float* W2    = (const float*)d_in[5];
  const float* b2    = (const float*)d_in[6];
  float* out = (float*)d_out;

  float* ws = (float*)d_ws;
  float* inner  = ws;                       // NV*NK = 10,000,000
  float* hid    = inner  + NV * NK;         // NB*NHID
  float* theta  = hid    + NB * NHID;       // NB*NK
  float* tnorm  = theta  + NB * NK;         // NB*NK
  float* colsum = tnorm  + NB * NK;         // NB*NK
  float* Abd    = colsum + NB * NK;         // NB*NK
  float* ntok   = Abd    + NB * NK;         // NB
  float* fdoc   = ntok   + NB;              // NB
  float* scale  = fdoc   + NB;              // 224 (padded)
  float* acc    = scale  + 224;             // 4

  k_init<<<dim3(2), dim3(256), 0, stream>>>(fdoc, scale, acc);
  k_ntok<<<dim3(NB), dim3(256), 0, stream>>>(bows, ntok);
  k_gemm_hid<<<dim3(NHID / 32, NB / 128), dim3(128), 0, stream>>>(bows, W1, b1, hid);
  k_theta<<<dim3((NB * NK + 255) / 256), dim3(256), 0, stream>>>(hid, W2, b2, theta);
  k_softmax_rows<<<dim3(NB), dim3(256), 0, stream>>>(theta, tnorm);
  k_gemm_inner<<<dim3(7, (NV + 127) / 128), dim3(128), 0, stream>>>(rho, alpha, inner);
  k_colred<<<dim3(NK), dim3(256), 0, stream>>>(inner, scale);
  k_fwd<<<dim3(NB / 64, (NV + 63) / 64), dim3(128), 0, stream>>>(inner, tnorm, bows, fdoc);
  k_colsumA<<<dim3((NK + 15) / 16, NB / 64), dim3(128), 0, stream>>>(bows, inner, colsum, Abd);
  k_recon<<<dim3((NV + 63) / 64, NB / 64), dim3(128), 0, stream>>>(theta, inner, scale, bows, acc);
  k_docfinal<<<dim3(NB), dim3(256), 0, stream>>>(Abd, colsum, tnorm, ntok, fdoc, acc);
  k_final<<<dim3(1), dim3(1), 0, stream>>>(acc, out);
  (void)in_sizes; (void)n_in; (void)out_size; (void)ws_size;
}
